// InteractionBlock_1451698946131
// MI455X (gfx1250) — compile-verified, matrix operand order checked by reading, not run on hardware
//
#include <hip/hip_runtime.h>
#include <hip/hip_bf16.h>
#include <cmath>

typedef _Float16 half_t;
typedef __attribute__((ext_vector_type(16))) _Float16 v16h;
typedef __attribute__((ext_vector_type(8)))  float    v8f;

static constexpr int   kHidden   = 128;
static constexpr int   kGauss    = 50;
static constexpr int   kGaussPad = 64;   // pad K to multiple of 32 for WMMA
static constexpr int   kFilt     = 128;
static constexpr int   kColors   = 4;
static constexpr float kCutoff   = 10.0f;
static constexpr float kPi       = 3.14159265358979323846f;
static constexpr float kLog2     = 0.6931471805599453f;

__device__ __forceinline__ float shifted_softplus(float v) {
  float sp = (v > 20.0f) ? v : log1pf(__expf(v));
  return sp - kLog2;
}

// ---------------- WMMA wave32 fragment helpers (ISA 7.12.2 layouts) ----------
// A: 16(M) x 32(K) fp16 in row-major LDS. lane L: row M = L&15; VGPR v holds
//   K = (v>>2)*16 + (L>>4)*8 + (v&3)*2  and K+1  (pairs contiguous -> b128).
__device__ __forceinline__ v16h load_frag_a(const half_t* sm, int ld) {
  const int lane = threadIdx.x & 31;
  const int m  = lane & 15;
  const int kh = (lane >> 4) * 8;
  v16h a;
#pragma unroll
  for (int v = 0; v < 8; ++v) {
    const int k0 = ((v >> 2) * 16) + kh + ((v & 3) * 2);
    a[2 * v]     = sm[m * ld + k0];
    a[2 * v + 1] = sm[m * ld + k0 + 1];
  }
  return a;
}

// B fragments are pre-swizzled in global memory: block of [32 lanes][16 halfs]
// per (fg, ks). Each lane reads its own contiguous 32B -> global_load_b128 x2.
__device__ __forceinline__ v16h load_frag_bp(const half_t* __restrict__ bf) {
  const int lane = threadIdx.x & 31;
  return *(const v16h*)(bf + lane * 16);
}

// 16xK fp16 LDS tile times packed-global Kx16 fp16 -> 16x16 f32.
// Two independent accumulators (even/odd k-steps) break the WMMA D->C chain so
// the XDL pipe can issue back-to-back and loads overlap the previous WMMA.
template <int K>
__device__ __forceinline__ v8f wave_gemm16p(const half_t* sA,
                                            const half_t* __restrict__ bF) {
  constexpr int NK = K / 32;
  v8f acc0 = {};
  v8f acc1 = {};
#pragma unroll
  for (int ks = 0; ks < NK; ks += 2) {
    v16h b0 = load_frag_bp(bF + ks * 512);
    v16h a0 = load_frag_a(sA + ks * 32, K);
    acc0 = __builtin_amdgcn_wmma_f32_16x16x32_f16(false, a0, false, b0,
                                                  (short)0, acc0, false, false);
    if (ks + 1 < NK) {
      v16h b1 = load_frag_bp(bF + (ks + 1) * 512);
      v16h a1 = load_frag_a(sA + (ks + 1) * 32, K);
      acc1 = __builtin_amdgcn_wmma_f32_16x16x32_f16(false, a1, false, b1,
                                                    (short)0, acc1, false,
                                                    false);
    }
  }
  if (NK > 1) {
#pragma unroll
    for (int i = 0; i < 8; ++i) acc0[i] += acc1[i];
  }
  return acc0;
}

// ------------------------------- kernels ------------------------------------

// Pre-swizzle a row-major fp32 weight matrix W[F_out, ldw] (valid K = kv,
// padded to n_ks*32) into fp16 WMMA-B fragment layout:
//   out[((fg*n_ks + ks)*32 + lane)*16 + j],  j=2v+hb,
//   k = ks*32 + (v>>2)*16 + (lane>>4)*8 + (v&3)*2 + hb,  f = fg*16 + (lane&15)
__global__ void pack_b_kernel(const float* __restrict__ W, int ldw, int kv,
                              int n_fg, int n_ks, half_t* __restrict__ out) {
  const int total = n_fg * n_ks * 512;
  for (int idx = blockIdx.x * blockDim.x + threadIdx.x; idx < total;
       idx += gridDim.x * blockDim.x) {
    const int j    = idx & 15;
    const int lane = (idx >> 4) & 31;
    const int ks   = (idx >> 9) % n_ks;
    const int fg   = idx / (n_ks * 512);
    const int v  = j >> 1;
    const int hb = j & 1;
    const int k =
        ks * 32 + ((v >> 2) * 16) + ((lane >> 4) * 8) + ((v & 3) * 2) + hb;
    const int f = fg * 16 + (lane & 15);
    out[idx] = (half_t)((k < kv) ? W[(size_t)f * ldw + k] : 0.0f);
  }
}

__global__ void init_kernel(float* agg, long long agg_n, int* sorted,
                            int sorted_n, int* counts) {
  const long long tid = (long long)blockIdx.x * blockDim.x + threadIdx.x;
  const long long stride = (long long)gridDim.x * blockDim.x;
  for (long long i = tid; i < agg_n; i += stride) agg[i] = 0.0f;
  for (long long i = tid; i < sorted_n; i += stride) sorted[i] = -1;
  if (tid < kColors) counts[tid] = 0;
}

__global__ void count_kernel(const int* __restrict__ colors, int n_edges,
                             int* counts) {
  const int e = blockIdx.x * blockDim.x + threadIdx.x;
  if (e < n_edges) atomicAdd(&counts[colors[e]], 1);
}

__global__ void offsets_kernel(const int* __restrict__ counts, int* cursor) {
  if (blockIdx.x == 0 && threadIdx.x == 0) {
    int off = 0;
    for (int c = 0; c < kColors; ++c) {
      cursor[c] = off;
      off += (counts[c] + 15) & ~15;  // 16-align each color segment
    }
  }
}

__global__ void scatter_kernel(const int* __restrict__ colors, int n_edges,
                               int* cursor, int* __restrict__ sorted) {
  const int e = blockIdx.x * blockDim.x + threadIdx.x;
  if (e < n_edges) {
    const int pos = atomicAdd(&cursor[colors[e]], 1);
    sorted[pos] = e;
  }
}

// h = x @ lin1_w.T — one wave per FULL 16-row tile (no bounds logic; the
// remainder rows are handled by lin1_tail_kernel).
__global__ void lin1_kernel(const float* __restrict__ x,
                            const half_t* __restrict__ p_lin1,
                            float* __restrict__ h) {
  __shared__ half_t sA[16 * kHidden];
  const int row0 = blockIdx.x * 16;
  const int lane = threadIdx.x;
  for (int idx = lane; idx < 16 * kHidden; idx += 32) {
    const int r = idx >> 7, k = idx & 127;
    sA[idx] = (half_t)x[(size_t)(row0 + r) * kHidden + k];
  }
  __syncthreads();
  const int ncol = lane & 15;
  const int mb = (lane >> 4) * 8;
  for (int fg = 0; fg < 8; ++fg) {
    v8f acc = wave_gemm16p<kHidden>(sA, p_lin1 + fg * 4 * 512);
    float* hp = h + (size_t)(row0 + mb) * kFilt + fg * 16 + ncol;
#pragma unroll
    for (int v = 0; v < 8; ++v) hp[v * kFilt] = acc[v];
  }
}

// Scalar tail for the last (<16) rows; never launched when N % 16 == 0.
__global__ void lin1_tail_kernel(const float* __restrict__ x,
                                 const float* __restrict__ lin1_w,
                                 float* __restrict__ h, int row_start,
                                 int n_nodes) {
  const int row = row_start + blockIdx.x;
  if (row >= n_nodes) return;
  const int f = threadIdx.x;
  float s = 0.0f;
  for (int k = 0; k < kHidden; ++k)
    s += x[(size_t)row * kHidden + k] * lin1_w[(size_t)f * kHidden + k];
  h[(size_t)row * kFilt + f] = s;
}

// One wave per 16-edge color-uniform tile: MLP1 -> ssp -> MLP2 -> modulate
// by cutoff and h[src], scatter-add into agg[dst] (L2-resident atomics).
// Biases/activations applied directly on WMMA accumulator registers.
// Padded rows have cw == 0, src/dst offset clamped to 0 -> msg == 0.0, so the
// scatter loop is branch-free.
__global__ void edge_kernel(const int* __restrict__ sorted,
                            const int* __restrict__ edge_index,
                            const float* __restrict__ edge_weight,
                            const float* __restrict__ edge_attr,
                            const int* __restrict__ colors,
                            const half_t* __restrict__ p_mlp1,
                            const float* __restrict__ mlp1_b,
                            const half_t* __restrict__ p_mlp2,
                            const float* __restrict__ mlp2_b,
                            const float* __restrict__ h,
                            float* __restrict__ agg, int n_edges) {
  __shared__ int    s_eid[16];
  __shared__ int    s_color;
  __shared__ float  s_cw[16];
  __shared__ half_t sA[16 * kGaussPad];
  __shared__ half_t sHh[16 * kFilt];

  const int base = blockIdx.x * 16;
  const int lane = threadIdx.x;
  if (lane < 16) {
    const int e = sorted[base + lane];
    s_eid[lane] = e;
    s_cw[lane] = (e >= 0)
        ? 0.5f * (cosf(edge_weight[e] * (kPi / kCutoff)) + 1.0f)
        : 0.0f;
  }
  if (lane == 0) {
    int c = -1;
    for (int i = 0; i < 16; ++i) {
      const int e = sorted[base + i];
      if (e >= 0) { c = colors[e]; break; }
    }
    s_color = c;
  }
  __syncthreads();
  const int c = s_color;
  if (c < 0) return;  // fully-padded tile (uniform branch, EXEC stays full)

  // Stage edge_attr tile, K padded 50 -> 64.
  for (int idx = lane; idx < 16 * kGaussPad; idx += 32) {
    const int r = idx >> 6, k = idx & 63;
    const int e = s_eid[r];
    sA[idx] = (half_t)((e >= 0 && k < kGauss)
                           ? edge_attr[(size_t)e * kGauss + k] : 0.0f);
  }
  __syncthreads();

  const int ncol = lane & 15;
  const int mb = (lane >> 4) * 8;

  // Preload this lane's 8 output-row metadata into registers (32-bit offsets;
  // n_nodes*128 < 2^31). Padded rows: cw = 0, offsets clamped to 0.
  float cw_r[8];
  int   srow[8], drow[8];
  const int* srcp = edge_index;
  const int* dstp = edge_index + n_edges;
#pragma unroll
  for (int v = 0; v < 8; ++v) {
    const int e = s_eid[mb + v];
    cw_r[v] = s_cw[mb + v];
    srow[v] = (e >= 0) ? srcp[e] * kFilt : 0;
    drow[v] = (e >= 0) ? dstp[e] * kFilt : 0;
  }

  // MLP layer 1: [16x64] @ [64x16] per filter group; bias+ssp on regs.
  const float*  b1  = mlp1_b + (size_t)c * kFilt;
  const half_t* bf1 = p_mlp1 + (size_t)c * 8 * 2 * 512;
  for (int fg = 0; fg < 8; ++fg) {
    v8f acc = wave_gemm16p<kGaussPad>(sA, bf1 + fg * 2 * 512);
    const int col = fg * 16 + ncol;
    const float bb = b1[col];
#pragma unroll
    for (int v = 0; v < 8; ++v)
      sHh[(mb + v) * kFilt + col] = (half_t)shifted_softplus(acc[v] + bb);
  }
  __syncthreads();

  // MLP layer 2 + cutoff * h[src] modulation + scatter-add (branch-free).
  const float*  b2  = mlp2_b + (size_t)c * kFilt;
  const half_t* bf2 = p_mlp2 + (size_t)c * 8 * 4 * 512;
  for (int fg = 0; fg < 8; ++fg) {
    v8f acc = wave_gemm16p<kFilt>(sHh, bf2 + fg * 4 * 512);
    const int col = fg * 16 + ncol;
    const float bb = b2[col];
#pragma unroll
    for (int v = 0; v < 8; ++v) {
      const float msg = (acc[v] + bb) * cw_r[v] * h[srow[v] + col];
      atomicAdd(&agg[drow[v] + col], msg);
    }
  }
}

// out = ssp(agg @ lin2_w.T + b2) @ lin_w.T + b — fused, one wave per FULL
// 16-row tile (remainder handled by out_tail_kernel).
__global__ void out_kernel(const float* __restrict__ agg,
                           const half_t* __restrict__ p_lin2,
                           const float* __restrict__ lin2_b,
                           const half_t* __restrict__ p_lin,
                           const float* __restrict__ lin_b,
                           float* __restrict__ out) {
  __shared__ half_t sA[16 * kFilt];
  __shared__ half_t sTh[16 * kHidden];
  const int row0 = blockIdx.x * 16;
  const int lane = threadIdx.x;
  for (int idx = lane; idx < 16 * kFilt; idx += 32) {
    const int r = idx >> 7, k = idx & 127;
    sA[idx] = (half_t)agg[(size_t)(row0 + r) * kFilt + k];
  }
  __syncthreads();
  const int ncol = lane & 15;
  const int mb = (lane >> 4) * 8;
  for (int fg = 0; fg < 8; ++fg) {
    v8f acc = wave_gemm16p<kFilt>(sA, p_lin2 + fg * 4 * 512);
    const int col = fg * 16 + ncol;
    const float bb = lin2_b[col];
#pragma unroll
    for (int v = 0; v < 8; ++v)
      sTh[(mb + v) * kHidden + col] = (half_t)shifted_softplus(acc[v] + bb);
  }
  __syncthreads();
  for (int fg = 0; fg < 8; ++fg) {
    v8f acc = wave_gemm16p<kHidden>(sTh, p_lin + fg * 4 * 512);
    const int col = fg * 16 + ncol;
    const float bb = lin_b[col];
    float* op = out + (size_t)(row0 + mb) * kHidden + col;
#pragma unroll
    for (int v = 0; v < 8; ++v) op[v * kHidden] = acc[v] + bb;
  }
}

// Scalar tail for the last (<16) rows; never launched when N % 16 == 0.
__global__ void out_tail_kernel(const float* __restrict__ agg,
                                const float* __restrict__ lin2_w,
                                const float* __restrict__ lin2_b,
                                const float* __restrict__ lin_w,
                                const float* __restrict__ lin_b,
                                float* __restrict__ out, int row_start,
                                int n_nodes) {
  __shared__ float t[kHidden];
  const int row = row_start + blockIdx.x;
  if (row >= n_nodes) return;
  const int f = threadIdx.x;
  float s = 0.0f;
  for (int k = 0; k < kFilt; ++k)
    s += agg[(size_t)row * kFilt + k] * lin2_w[(size_t)f * kFilt + k];
  t[f] = shifted_softplus(s + lin2_b[f]);
  __syncthreads();
  float o = 0.0f;
  for (int k = 0; k < kHidden; ++k)
    o += t[k] * lin_w[(size_t)f * kHidden + k];
  out[(size_t)row * kHidden + f] = o + lin_b[f];
}

// ------------------------------ launcher ------------------------------------

extern "C" void kernel_launch(void* const* d_in, const int* in_sizes, int n_in,
                              void* d_out, int out_size, void* d_ws,
                              size_t ws_size, hipStream_t stream) {
  const float* x           = (const float*)d_in[0];
  const int*   edge_index  = (const int*)d_in[1];
  const float* edge_weight = (const float*)d_in[2];
  const float* edge_attr   = (const float*)d_in[3];
  const int*   colors      = (const int*)d_in[4];
  const float* mlp1_w      = (const float*)d_in[5];
  const float* mlp1_b      = (const float*)d_in[6];
  const float* mlp2_w      = (const float*)d_in[7];
  const float* mlp2_b      = (const float*)d_in[8];
  const float* lin1_w      = (const float*)d_in[9];
  const float* lin2_w      = (const float*)d_in[10];
  const float* lin2_b      = (const float*)d_in[11];
  const float* lin_w       = (const float*)d_in[12];
  const float* lin_b       = (const float*)d_in[13];
  float* out = (float*)d_out;

  const int n_nodes = in_sizes[0] / kHidden;
  const int n_edges = in_sizes[2];
  const int n_full  = n_nodes / 16;          // full WMMA tiles
  const int tail0   = n_full * 16;           // first remainder row
  const int n_tail  = n_nodes - tail0;       // 0 for N % 16 == 0
  // Per-color segments 16-aligned: worst case n_edges + (kColors-1)*15 slots.
  const int sorted_n = (n_edges + (kColors - 1) * 15 + 15) & ~15;

  size_t off = 0;
  auto take = [&](size_t bytes) {
    void* p = (char*)d_ws + off;
    off += (bytes + 255) & ~(size_t)255;
    return p;
  };
  float*  agg    = (float*)take((size_t)n_nodes * kFilt * sizeof(float));
  float*  h      = (float*)take((size_t)n_nodes * kFilt * sizeof(float));
  int*    sorted = (int*)take((size_t)sorted_n * sizeof(int));
  int*    counts = (int*)take(kColors * sizeof(int));
  int*    cursor = (int*)take(kColors * sizeof(int));
  // Packed fp16 B-fragment weights.
  half_t* p_mlp1 = (half_t*)take((size_t)kColors * 8 * 2 * 512 * sizeof(half_t));
  half_t* p_mlp2 = (half_t*)take((size_t)kColors * 8 * 4 * 512 * sizeof(half_t));
  half_t* p_lin1 = (half_t*)take((size_t)8 * 4 * 512 * sizeof(half_t));
  half_t* p_lin2 = (half_t*)take((size_t)8 * 4 * 512 * sizeof(half_t));
  half_t* p_lin  = (half_t*)take((size_t)8 * 4 * 512 * sizeof(half_t));
  (void)ws_size; (void)n_in; (void)out_size;

  // Weight pre-swizzle (tiny; L2-resident afterwards). mlp1/mlp2 treated as
  // [4*128, K] row-major so color c's fragments sit at fg = c*8 .. c*8+7.
  pack_b_kernel<<<128, 256, 0, stream>>>(mlp1_w, kGauss, kGauss, 32, 2, p_mlp1);
  pack_b_kernel<<<256, 256, 0, stream>>>(mlp2_w, kFilt, kFilt, 32, 4, p_mlp2);
  pack_b_kernel<<<64, 256, 0, stream>>>(lin1_w, kHidden, kHidden, 8, 4, p_lin1);
  pack_b_kernel<<<64, 256, 0, stream>>>(lin2_w, kFilt, kFilt, 8, 4, p_lin2);
  pack_b_kernel<<<64, 256, 0, stream>>>(lin_w, kHidden, kHidden, 8, 4, p_lin);

  const long long agg_n = (long long)n_nodes * kFilt;
  init_kernel<<<1024, 256, 0, stream>>>(agg, agg_n, sorted, sorted_n, counts);
  count_kernel<<<(n_edges + 255) / 256, 256, 0, stream>>>(colors, n_edges,
                                                          counts);
  offsets_kernel<<<1, 32, 0, stream>>>(counts, cursor);
  scatter_kernel<<<(n_edges + 255) / 256, 256, 0, stream>>>(colors, n_edges,
                                                            cursor, sorted);
  if (n_full > 0)
    lin1_kernel<<<n_full, 32, 0, stream>>>(x, p_lin1, h);
  if (n_tail > 0)
    lin1_tail_kernel<<<n_tail, kFilt, 0, stream>>>(x, lin1_w, h, tail0,
                                                   n_nodes);
  edge_kernel<<<sorted_n / 16, 32, 0, stream>>>(
      sorted, edge_index, edge_weight, edge_attr, colors, p_mlp1, mlp1_b,
      p_mlp2, mlp2_b, h, agg, n_edges);
  if (n_full > 0)
    out_kernel<<<n_full, 32, 0, stream>>>(agg, p_lin2, lin2_b, p_lin, lin_b,
                                          out);
  if (n_tail > 0)
    out_tail_kernel<<<n_tail, kHidden, 0, stream>>>(agg, lin2_w, lin2_b,
                                                    lin_w, lin_b, out, tail0,
                                                    n_nodes);
}
